// GATPoolSep_43593918054682
// MI455X (gfx1250) — matrix operand order repaired and checked by reading, not compiled, not run on hardware
//
#include <hip/hip_runtime.h>
#include <hip/hip_bf16.h>

// ---------------------------------------------------------------------------
// GAT layer for MI455X (gfx1250): bf16 WMMA GEMMs + CSR-by-dst edge softmax
// ---------------------------------------------------------------------------

typedef __attribute__((ext_vector_type(16))) __bf16 v16bf;
typedef __attribute__((ext_vector_type(8)))  __bf16 v8bf;
typedef __attribute__((ext_vector_type(8)))  float  v8f;

#define N_NODES 50000
#define N_EDGES 1000000
#define F_INP   256
#define HEADS   8
#define DHEAD   64
#define HDIM    512   // HEADS*DHEAD

// ---------------------------------------------------------------------------
// Packing kernels: fp32 -> bf16, weight transposes
// ---------------------------------------------------------------------------
__global__ void pack_bf16(const float* __restrict__ src, __bf16* __restrict__ dst, int n) {
    int i = blockIdx.x * blockDim.x + threadIdx.x;
    if (i < n) dst[i] = (__bf16)src[i];
}

// W [H, F, D] -> WcatT [C=H*D, F] bf16   (column c = h*64+d of the concat projection)
__global__ void pack_wcatT(const float* __restrict__ W, __bf16* __restrict__ Wt) {
    int i = blockIdx.x * blockDim.x + threadIdx.x;          // HDIM*F_INP
    int f = i & (F_INP - 1);
    int c = i >> 8;
    int h = c >> 6, d = c & 63;
    Wt[(size_t)c * F_INP + f] =
        (__bf16)W[(size_t)h * F_INP * DHEAD + (size_t)f * DHEAD + d];
}

// hidden_w [in=512, out=256] -> Ht [out, in] bf16
__global__ void pack_hT(const float* __restrict__ Hw, __bf16* __restrict__ Ht) {
    int idx = blockIdx.x * blockDim.x + threadIdx.x;        // 256*512
    int i = idx & (HDIM - 1);
    int o = idx >> 9;
    Ht[(size_t)o * HDIM + i] = (__bf16)Hw[(size_t)i * F_INP + o];
}

// lin_w [in=256, out=256] -> Lt [out, in] bf16
__global__ void pack_lT(const float* __restrict__ Lw, __bf16* __restrict__ Lt) {
    int idx = blockIdx.x * blockDim.x + threadIdx.x;        // 256*256
    int i = idx & (F_INP - 1);
    int o = idx >> 8;
    Lt[(size_t)o * F_INP + i] = (__bf16)Lw[(size_t)i * F_INP + o];
}

// ---------------------------------------------------------------------------
// WMMA GEMM:  C[M x Nout] = A[M,K](bf16,row-major) * Bt[Nout,K](bf16,row-major)^T
// One wave32 per 16x16 tile; blockDim = (32, 8) -> 8 column tiles per block.
// MODE 0: store fp32          MODE 1: +bias, store bf16      MODE 2: +resid, fp32
// Fragment layouts per CDNA5 ISA 7.12.2:
//   A (16x32 bf16): lane l -> row l&15, k = (l>>4)*8 + {0..7} and +16+{0..7}
//   B (32x16 bf16): lane l -> col l&15, k = (l>>4)*8 + {0..7} and +16+{0..7}
//   C/D fp32:       lane l, vgpr v -> row v + (l>>4)*8, col l&15
// ---------------------------------------------------------------------------
template <int MODE>
__global__ void wmma_gemm(const __bf16* __restrict__ A, const __bf16* __restrict__ Bt,
                          int K, int Nout,
                          const float* __restrict__ bias,
                          const float* __restrict__ resid,
                          float* __restrict__ outF,
                          __bf16* __restrict__ outB) {
    const int lane = threadIdx.x;
    const int half = lane >> 4;
    const int l16  = lane & 15;
    const int row0 = blockIdx.x * 16;
    const int col0 = (blockIdx.y * blockDim.y + threadIdx.y) * 16;

    const __bf16* Arow = A  + (size_t)(row0 + l16) * K;
    const __bf16* Brow = Bt + (size_t)(col0 + l16) * K;

    v8f acc = {};
    for (int kc = 0; kc < K; kc += 32) {
        v8bf alo = *(const v8bf*)(Arow + kc + half * 8);
        v8bf ahi = *(const v8bf*)(Arow + kc + half * 8 + 16);
        v16bf af = __builtin_shufflevector(alo, ahi,
                       0, 1, 2, 3, 4, 5, 6, 7, 8, 9, 10, 11, 12, 13, 14, 15);
        v8bf blo = *(const v8bf*)(Brow + kc + half * 8);
        v8bf bhi = *(const v8bf*)(Brow + kc + half * 8 + 16);
        v16bf bf = __builtin_shufflevector(blo, bhi,
                       0, 1, 2, 3, 4, 5, 6, 7, 8, 9, 10, 11, 12, 13, 14, 15);
        acc = __builtin_amdgcn_wmma_f32_16x16x32_bf16(
                  false, af, false, bf, (short)0, acc, false, false);
    }

    const int col = col0 + l16;
#pragma unroll
    for (int v = 0; v < 8; v++) {
        const int row = row0 + v + half * 8;
        const size_t idx = (size_t)row * Nout + col;
        float val = acc[v];
        if constexpr (MODE == 0) {
            outF[idx] = val;
        } else if constexpr (MODE == 1) {
            outB[idx] = (__bf16)(val + bias[col]);
        } else {
            outF[idx] = val + resid[idx];
        }
    }
}

// ---------------------------------------------------------------------------
// Per-(node, head) attention halves: s_src/s_dst [N*H], dot(Wh_row, a_half)
// ---------------------------------------------------------------------------
__global__ void scores_kernel(const float* __restrict__ Wh, const float* __restrict__ a,
                              float* __restrict__ ss, float* __restrict__ sd) {
    int idx = blockIdx.x * blockDim.x + threadIdx.x;
    if (idx >= N_NODES * HEADS) return;
    int n = idx >> 3, h = idx & 7;
    const float* wr = Wh + (size_t)n * HDIM + h * DHEAD;
    const float* av = a + h * (2 * DHEAD);
    float s0 = 0.f, s1 = 0.f;
#pragma unroll 8
    for (int d = 0; d < DHEAD; d++) {
        float w = wr[d];
        s0 = fmaf(w, av[d], s0);
        s1 = fmaf(w, av[DHEAD + d], s1);
    }
    ss[idx] = s0;
    sd[idx] = s1;
}

// ---------------------------------------------------------------------------
// CSR-by-destination construction
// ---------------------------------------------------------------------------
__global__ void zero_i32(int* __restrict__ p, int n) {
    int i = blockIdx.x * blockDim.x + threadIdx.x;
    if (i < n) p[i] = 0;
}

__global__ void count_kernel(const int* __restrict__ ei, int* __restrict__ counts) {
    int i = blockIdx.x * blockDim.x + threadIdx.x;
    if (i < N_EDGES) atomicAdd(&counts[ei[N_EDGES + i]], 1);
}

// single-block 1024-thread exclusive scan over counts[N] -> rowstart, cursor
__global__ void scan_kernel(const int* __restrict__ counts, int* __restrict__ rowstart,
                            int* __restrict__ cursor, int n) {
    __shared__ int buf[1024];
    __shared__ int carrySh;
    const int t = threadIdx.x;
    if (t == 0) carrySh = 0;
    for (int base = 0; base < n; base += 1024) {
        __syncthreads();
        const int carry = carrySh;
        const int v = (base + t < n) ? counts[base + t] : 0;
        buf[t] = v;
        __syncthreads();
        for (int off = 1; off < 1024; off <<= 1) {
            int y = (t >= off) ? buf[t - off] : 0;
            __syncthreads();
            buf[t] += y;
            __syncthreads();
        }
        const int incl = buf[t];
        const int excl = carry + incl - v;
        if (base + t < n) { rowstart[base + t] = excl; cursor[base + t] = excl; }
        __syncthreads();
        if (t == 1023) carrySh = carry + incl;
    }
}

__global__ void fill_kernel(const int* __restrict__ ei, int* __restrict__ cursor,
                            int* __restrict__ csr_src) {
    int i = blockIdx.x * blockDim.x + threadIdx.x;
    if (i < N_EDGES) {
        int d = ei[N_EDGES + i];
        int pos = atomicAdd(&cursor[d], 1);
        csr_src[pos] = ei[i];
    }
}

// ---------------------------------------------------------------------------
// Aggregation: one block per dst node; one wave32 per head.
// Softmax over incoming edges with wave-shuffle reductions (no global atomics),
// weighted gather of Wh[src] rows, ELU, store bf16 [N, H*D] for the next GEMM.
// ---------------------------------------------------------------------------
__global__ __launch_bounds__(256) void agg_kernel(
        const int* __restrict__ rowstart, const int* __restrict__ counts,
        const int* __restrict__ csr_src,
        const float* __restrict__ ss, const float* __restrict__ sd,
        const float* __restrict__ Wh, __bf16* __restrict__ outb) {
    const int n    = blockIdx.x;
    const int h    = threadIdx.y;     // 0..7
    const int lane = threadIdx.x;     // 0..31
    const int start = rowstart[n];
    const int deg   = counts[n];
    const float sdn = sd[n * HEADS + h];

    // pass 1: max of leaky-relu scores
    float m = -3.0e38f;
    for (int base = 0; base < deg; base += 32) {
        int e = base + lane;
        float v = -3.0e38f;
        if (e < deg) {
            int s = csr_src[start + e];
            float t = ss[s * HEADS + h] + sdn;
            v = t > 0.f ? t : 0.2f * t;
        }
        m = fmaxf(m, v);
    }
    for (int off = 16; off; off >>= 1) m = fmaxf(m, __shfl_xor(m, off, 32));

    // pass 2: softmax denominator
    float sum = 0.f;
    for (int base = 0; base < deg; base += 32) {
        int e = base + lane;
        if (e < deg) {
            int s = csr_src[start + e];
            float t = ss[s * HEADS + h] + sdn;
            t = t > 0.f ? t : 0.2f * t;
            sum += __expf(t - m);
        }
    }
    for (int off = 16; off; off >>= 1) sum += __shfl_xor(sum, off, 32);
    const float inv = 1.0f / (sum + 1e-16f);

    // pass 3: weighted aggregation; lane owns dims {lane, lane+32}
    float acc0 = 0.f, acc1 = 0.f;
    for (int base = 0; base < deg; base += 32) {
        const int cnt = min(32, deg - base);
        int e = base + lane;
        float att = 0.f;
        int sE = 0;
        if (e < deg) {
            sE = csr_src[start + e];
            float t = ss[sE * HEADS + h] + sdn;
            t = t > 0.f ? t : 0.2f * t;
            att = __expf(t - m) * inv;
        }
        for (int j = 0; j < cnt; j++) {
            float aj = __shfl(att, j, 32);
            int   sj = __shfl(sE,  j, 32);
            const float* wr = Wh + (size_t)sj * HDIM + h * DHEAD;
            acc0 = fmaf(aj, wr[lane],      acc0);
            acc1 = fmaf(aj, wr[lane + 32], acc1);
        }
    }

    // ELU (concat=True) then store bf16 for the hidden GEMM
    acc0 = acc0 > 0.f ? acc0 : (__expf(acc0) - 1.f);
    acc1 = acc1 > 0.f ? acc1 : (__expf(acc1) - 1.f);
    const size_t o = (size_t)n * HDIM + h * DHEAD;
    outb[o + lane]      = (__bf16)acc0;
    outb[o + 32 + lane] = (__bf16)acc1;
}

// ---------------------------------------------------------------------------
// Launch
// ---------------------------------------------------------------------------
extern "C" void kernel_launch(void* const* d_in, const int* in_sizes, int n_in,
                              void* d_out, int out_size, void* d_ws, size_t ws_size,
                              hipStream_t stream) {
    const float* x   = (const float*)d_in[0];
    const int*   ei  = (const int*)d_in[1];     // [2, E] int32
    const float* W   = (const float*)d_in[2];   // [H, F, D]
    const float* a   = (const float*)d_in[3];   // [H, 2D]
    const float* hw  = (const float*)d_in[4];   // [H*D, F]
    const float* hb  = (const float*)d_in[5];   // [F]
    const float* lw  = (const float*)d_in[6];   // [F, F]
    float* out = (float*)d_out;                 // [N, F]

    // workspace carve-out (256B aligned)
    char* p = (char*)d_ws;
    auto carve = [&](size_t bytes) -> void* {
        void* r = (void*)p;
        p += (bytes + 255) & ~(size_t)255;
        return r;
    };
    float*  Wh     = (float*) carve((size_t)N_NODES * HDIM * 4);   // 102.4 MB
    __bf16* xb     = (__bf16*)carve((size_t)N_NODES * F_INP * 2);  //  25.6 MB
    __bf16* outb   = (__bf16*)carve((size_t)N_NODES * HDIM * 2);   //  51.2 MB
    __bf16* h1b    = (__bf16*)carve((size_t)N_NODES * F_INP * 2);  //  25.6 MB
    __bf16* WcatT  = (__bf16*)carve((size_t)HDIM * F_INP * 2);
    __bf16* hT     = (__bf16*)carve((size_t)F_INP * HDIM * 2);
    __bf16* lT     = (__bf16*)carve((size_t)F_INP * F_INP * 2);
    float*  ssrc   = (float*) carve((size_t)N_NODES * HEADS * 4);
    float*  sdst   = (float*) carve((size_t)N_NODES * HEADS * 4);
    int*    counts = (int*)   carve((size_t)N_NODES * 4);
    int*    rowst  = (int*)   carve((size_t)N_NODES * 4);
    int*    cursor = (int*)   carve((size_t)N_NODES * 4);
    int*    csrsrc = (int*)   carve((size_t)N_EDGES * 4);
    (void)ws_size; (void)in_sizes; (void)n_in; (void)out_size;

    // 1) pack inputs / weights to bf16
    pack_bf16<<<(N_NODES * F_INP + 255) / 256, 256, 0, stream>>>(x, xb, N_NODES * F_INP);
    pack_wcatT<<<(HDIM * F_INP + 255) / 256, 256, 0, stream>>>(W, WcatT);
    pack_hT<<<(F_INP * HDIM + 255) / 256, 256, 0, stream>>>(hw, hT);
    pack_lT<<<(F_INP * F_INP + 255) / 256, 256, 0, stream>>>(lw, lT);

    // 2) Wh = x @ Wcat   (50000 x 256 x 512), fp32 out
    {
        dim3 blk(32, 8), grd(N_NODES / 16, HDIM / (16 * 8));
        wmma_gemm<0><<<grd, blk, 0, stream>>>(xb, WcatT, F_INP, HDIM,
                                              nullptr, nullptr, Wh, nullptr);
    }

    // 3) attention halves
    scores_kernel<<<(N_NODES * HEADS + 255) / 256, 256, 0, stream>>>(Wh, a, ssrc, sdst);

    // 4) CSR by destination
    zero_i32<<<(N_NODES + 255) / 256, 256, 0, stream>>>(counts, N_NODES);
    count_kernel<<<(N_EDGES + 255) / 256, 256, 0, stream>>>(ei, counts);
    scan_kernel<<<1, 1024, 0, stream>>>(counts, rowst, cursor, N_NODES);
    fill_kernel<<<(N_EDGES + 255) / 256, 256, 0, stream>>>(ei, cursor, csrsrc);

    // 5) softmax + aggregation + ELU -> bf16 [N, H*D]
    {
        dim3 blk(32, 8);
        agg_kernel<<<N_NODES, blk, 0, stream>>>(rowst, counts, csrsrc,
                                                ssrc, sdst, Wh, outb);
    }

    // 6) h1 = ELUed @ hidden_w + b   (50000 x 512 x 256), bf16 out
    {
        dim3 blk(32, 8), grd(N_NODES / 16, F_INP / (16 * 8));
        wmma_gemm<1><<<grd, blk, 0, stream>>>(outb, hT, HDIM, F_INP,
                                              hb, nullptr, nullptr, h1b);
    }

    // 7) out = h1 @ lin_w + x   (50000 x 256 x 256), fp32 out with residual
    {
        dim3 blk(32, 8), grd(N_NODES / 16, F_INP / (16 * 8));
        wmma_gemm<2><<<grd, blk, 0, stream>>>(h1b, lT, F_INP, F_INP,
                                              nullptr, x, out, nullptr);
    }
}
